// SDGRATModel_18442589569184
// MI455X (gfx1250) — matrix operand order repaired and checked by reading, not compiled, not run on hardware
//
#include <hip/hip_runtime.h>
#include <hip/hip_bf16.h>

// ---------------------------------------------------------------------------
// Fused SDGRAT model for MI455X (gfx1250), wave32 + v_wmma_f32_16x16x32_f16.
// One workgroup (128 threads = 4 waves) per batch graph; all activations in
// LDS; weights pre-converted to f16 in d_ws and streamed from L2.
//
// NOTE on the GCN blocks: the reference's edge_index.repeat(B,1,1).view(2,-1)
// quirk makes src[i]==dst[i] for every edge, so the symmetric-norm GCN
// aggregation collapses exactly to out = h + gcn_b for every node. Both GCN
// blocks are therefore relu(LN(x@W^T + b + res)) -- implemented that way.
// ---------------------------------------------------------------------------

#define J    28
#define JP   32
#define DD   128
#define FFD  256
#define NH   4
#define DH   32
#define OUTD 3
#define QKV  384

typedef __attribute__((ext_vector_type(16))) _Float16 v16h;
typedef __attribute__((ext_vector_type(8)))  float    v8f;

union H16 { v16h v; _Float16 h[16]; };

// ---- WMMA fragment loaders (contiguous k = 16*(lane/16) + s, wave32) -------------------

// A fragment 16x32 f16: lanes 0-15 -> K=0-15, lanes 16-31 -> K=16-31;
// VGPR v: K=kb+2v{,+1}.  p -> A[row0][k0].
__device__ inline v16h load_a_frag(const _Float16* p, int lda, int lane) {
  const int m  = lane & 15;
  const int kb = (lane >> 4) << 4;           // 0 or 16
  const _Float16* row = p + m * lda + kb;
  H16 r;
#pragma unroll
  for (int v = 0; v < 8; ++v) {
    r.h[2 * v]     = row[2 * v];
    r.h[2 * v + 1] = row[2 * v + 1];
  }
  return r.v;
}

// B fragment 32x16 with B[k][n] = W[n][k], W row-major N x K. p -> W[n0][k0].
// lanes 0-15: K=0-15 (VGPR v -> K=2v,2v+1); lanes 16-31: K=16-31.
__device__ inline v16h load_b_nk(const _Float16* p, int ldw, int lane) {
  const int n  = lane & 15;
  const int kb = (lane >> 4) << 4;           // 0 or 16
  const _Float16* row = p + n * ldw + kb;
  H16 r;
#pragma unroll
  for (int v = 0; v < 8; ++v) {
    r.h[2 * v]     = row[2 * v];
    r.h[2 * v + 1] = row[2 * v + 1];
  }
  return r.v;
}

// B fragment 32x16 with B stored row-major K x N (used for P @ V). p -> B[k0][n0].
__device__ inline v16h load_b_kn(const _Float16* p, int ldb, int lane) {
  const int n  = lane & 15;
  const int kb = (lane >> 4) << 4;
  H16 r;
#pragma unroll
  for (int v = 0; v < 8; ++v) {
    r.h[2 * v]     = p[(kb + 2 * v) * ldb + n];
    r.h[2 * v + 1] = p[(kb + 2 * v + 1) * ldb + n];
  }
  return r.v;
}

// C = A(32xK, lda=QKV, f16 LDS) @ W^T (W: NxK f16 row-major) + bias, C f32 LDS.
// RELU is compile-time; bias is unconditional (all call sites have one) so the
// inner loop carries no data-dependent branches.
template <bool RELU>
__device__ inline void gemm_xwt(const _Float16* A, const _Float16* W, int K, int N,
                                float* C, int ldc, const float* __restrict__ bias,
                                int wave, int lane) {
  const int ntc = N >> 4;
  for (int nt = wave; nt < ntc; nt += 4) {
    v8f acc0 = {}; v8f acc1 = {};
    for (int kt = 0; kt < (K >> 5); ++kt) {
      v16h bf = load_b_nk(W + nt * 16 * K + kt * 32, K, lane);
      v16h a0 = load_a_frag(A + kt * 32, QKV, lane);
      v16h a1 = load_a_frag(A + 16 * QKV + kt * 32, QKV, lane);
      acc0 = __builtin_amdgcn_wmma_f32_16x16x32_f16(false, a0, false, bf, (short)0, acc0, false, false);
      acc1 = __builtin_amdgcn_wmma_f32_16x16x32_f16(false, a1, false, bf, (short)0, acc1, false, false);
    }
    const int n   = nt * 16 + (lane & 15);
    const float bv = bias[n];
    const int mb  = (lane >> 4) << 3;
#pragma unroll
    for (int v = 0; v < 8; ++v) {
      float y0 = acc0[v] + bv, y1 = acc1[v] + bv;
      if (RELU) { y0 = fmaxf(y0, 0.0f); y1 = fmaxf(y1, 0.0f); }
      C[(mb + v) * ldc + n]      = y0;
      C[(16 + mb + v) * ldc + n] = y1;
    }
  }
}

// LayerNorm over D=128: z = In (+Res) (+bias); y = LN(z)*g + b (opt relu).
// Writes y to resOut (f32, ld 128) and A16 cols 0..127 (lda QKV).
template <bool HAVE_RES, bool HAVE_BIAS, bool RELU>
__device__ inline void layernorm(const float* In, const float* Res, const float* bias,
                                 const float* g, const float* bt,
                                 float* resOut, _Float16* aOut, int wave, int lane) {
  for (int m = wave; m < J; m += 4) {
    float vbuf[4]; float s = 0.f, s2 = 0.f;
#pragma unroll
    for (int i = 0; i < 4; ++i) {
      const int d = lane + 32 * i;
      float z = In[m * DD + d];
      if (HAVE_RES)  z += Res[m * DD + d];
      if (HAVE_BIAS) z += bias[d];
      vbuf[i] = z; s += z; s2 += z * z;
    }
#pragma unroll
    for (int off = 16; off > 0; off >>= 1) {
      s  += __shfl_xor(s, off, 32);
      s2 += __shfl_xor(s2, off, 32);
    }
    const float mean = s * (1.0f / 128.0f);
    const float var  = s2 * (1.0f / 128.0f) - mean * mean;
    const float rstd = rsqrtf(var + 1e-5f);
#pragma unroll
    for (int i = 0; i < 4; ++i) {
      const int d = lane + 32 * i;
      float y = (vbuf[i] - mean) * rstd * g[d] + bt[d];
      if (RELU) y = fmaxf(y, 0.0f);
      resOut[m * DD + d] = y;
      aOut[m * QKV + d]  = (_Float16)y;
    }
  }
}

struct Params {
  const float* x;
  const float *g1W, *g1b, *g1lng, *g1lnb, *rW, *rb;
  const _Float16* g2W16; const float *g2b, *g2lng, *g2lnb;
  const float *pos, *rout;
  const _Float16* inW16[2];  const float* inb[2];
  const _Float16* outW16[2]; const float* outb[2];
  const float *ln1g[2], *ln1b[2];
  const _Float16* ff1W16[2]; const float* ff1b[2];
  const _Float16* ff2W16[2]; const float* ff2b[2];
  const float *ln2g[2], *ln2b[2];
  const float *flng, *flnb;
  const _Float16* dW1_16; const float *db1, *dW2, *db2;
  float* out;
};

__global__ void __launch_bounds__(128) f32_to_f16_kernel(const float* src, _Float16* dst, int n) {
  int i = blockIdx.x * 128 + threadIdx.x;
  if (i < n) dst[i] = (_Float16)src[i];
}

__global__ void __launch_bounds__(128) sdgrat_fused(Params P) {
  __shared__ _Float16 A16[JP * QKV];     // f16 activation (A operand), lda=QKV
  __shared__ float    C32[JP * QKV];     // f32 GEMM accumulators
  __shared__ float    XRES[JP * DD];     // f32 residual stream
  __shared__ float    S32[NH * JP * JP]; // attention scores
  __shared__ _Float16 P16[NH * JP * JP]; // attention probs (f16)
  __shared__ float    xl[J * 2];

  const int b    = blockIdx.x;
  const int tid  = threadIdx.x;
  const int wave = tid >> 5;
  const int lane = tid & 31;

  // zero A16 pad rows 28..31 (kept zero by all later converts)
  for (int i = tid; i < (JP - J) * QKV; i += 128) A16[J * QKV + i] = (_Float16)0.f;
  if (tid < J * 2) xl[tid] = P.x[b * (J * 2) + tid];
  __syncthreads();

  // ---- GCN1 (K=2, scalar): h = x@W^T ; res = x@resW^T + resb ----
  for (int idx = tid; idx < J * DD; idx += 128) {
    const int m = idx >> 7, d = idx & 127;
    const float x0 = xl[2 * m], x1 = xl[2 * m + 1];
    C32[m * DD + d]  = x0 * P.g1W[2 * d] + x1 * P.g1W[2 * d + 1];
    XRES[m * DD + d] = x0 * P.rW[2 * d]  + x1 * P.rW[2 * d + 1] + P.rb[d];
  }
  __syncthreads();
  // out = h + gcn_b (self-loop collapse) + res -> LN -> relu
  layernorm<true, true, true>(C32, XRES, P.g1b, P.g1lng, P.g1lnb, XRES, A16, wave, lane);
  __syncthreads();

  // ---- GCN2: h2 = h1@W^T + b; out = h2 + h1 -> LN -> relu ----
  gemm_xwt<false>(A16, P.g2W16, DD, DD, C32, DD, P.g2b, wave, lane);
  __syncthreads();
  layernorm<true, false, true>(C32, XRES, nullptr, P.g2lng, P.g2lnb, XRES, A16, wave, lane);
  __syncthreads();

  // ---- + pos_embed + routing_bias ----
  for (int idx = tid; idx < J * DD; idx += 128) {
    const int m = idx >> 7, d = idx & 127;
    const float v = XRES[idx] + P.pos[idx] + P.rout[idx];
    XRES[idx] = v;
    A16[m * QKV + d] = (_Float16)v;
  }
  __syncthreads();

  // ---- 2 transformer encoder layers (post-norm) ----
  for (int l = 0; l < 2; ++l) {
    // QKV projection: (28,128)@(128,384)
    gemm_xwt<false>(A16, P.inW16[l], DD, QKV, C32, QKV, P.inb[l], wave, lane);
    __syncthreads();
    // convert qkv f32 -> f16 (overwrites x in A16; x f32 kept in XRES)
    for (int idx = tid; idx < JP * QKV; idx += 128) {
      const int m = idx / QKV;
      A16[idx] = (_Float16)((m < J) ? C32[idx] : 0.0f);
    }
    __syncthreads();

    // scores: S[h] = Q_h @ K_h^T  (K=32, 2 M-tiles x 2 N-tiles x 4 heads)
    for (int u = wave; u < 8; u += 4) {
      const int h = u >> 1, nt = u & 1;
      v16h bf = load_b_nk(A16 + (nt * 16) * QKV + DD + DH * h, QKV, lane);
      v16h a0 = load_a_frag(A16 + DH * h, QKV, lane);
      v16h a1 = load_a_frag(A16 + 16 * QKV + DH * h, QKV, lane);
      v8f acc0 = {}, acc1 = {};
      acc0 = __builtin_amdgcn_wmma_f32_16x16x32_f16(false, a0, false, bf, (short)0, acc0, false, false);
      acc1 = __builtin_amdgcn_wmma_f32_16x16x32_f16(false, a1, false, bf, (short)0, acc1, false, false);
      float* Sh = S32 + h * JP * JP;
      const int n  = nt * 16 + (lane & 15);
      const int mb = (lane >> 4) << 3;
#pragma unroll
      for (int v = 0; v < 8; ++v) {
        Sh[(mb + v) * JP + n]      = acc0[v];
        Sh[(16 + mb + v) * JP + n] = acc1[v];
      }
    }
    __syncthreads();

    // softmax: 4 heads * 28 rows = 112 rows over 128 threads
    if (tid < NH * J) {
      const int h = tid / J, m = tid % J;
      float* srow = S32 + h * JP * JP + m * JP;
      const float scale = 0.17677669529663687f;  // 1/sqrt(32)
      float mx = -1e30f;
      for (int n = 0; n < J; ++n) { const float v = srow[n] * scale; srow[n] = v; mx = fmaxf(mx, v); }
      float s = 0.f;
      for (int n = 0; n < J; ++n) { const float e = __expf(srow[n] - mx); srow[n] = e; s += e; }
      const float inv = 1.0f / s;
      _Float16* prow = P16 + h * JP * JP + m * JP;
      for (int n = 0; n < J; ++n)  prow[n] = (_Float16)(srow[n] * inv);
      for (int n = J; n < JP; ++n) prow[n] = (_Float16)0.f;
    } else if (tid < NH * J + 16) {
      const int u = tid - NH * J;
      _Float16* prow = P16 + (u >> 2) * JP * JP + (J + (u & 3)) * JP;
      for (int n = 0; n < JP; ++n) prow[n] = (_Float16)0.f;
    }
    __syncthreads();

    // O = P @ V  -> C32 viewed as (32 x 128)
    for (int u = wave; u < 8; u += 4) {
      const int h = u >> 1, nt = u & 1;
      v16h bf = load_b_kn(A16 + 2 * DD + DH * h + nt * 16, QKV, lane);
      v16h a0 = load_a_frag(P16 + h * JP * JP, JP, lane);
      v16h a1 = load_a_frag(P16 + h * JP * JP + 16 * JP, JP, lane);
      v8f acc0 = {}, acc1 = {};
      acc0 = __builtin_amdgcn_wmma_f32_16x16x32_f16(false, a0, false, bf, (short)0, acc0, false, false);
      acc1 = __builtin_amdgcn_wmma_f32_16x16x32_f16(false, a1, false, bf, (short)0, acc1, false, false);
      const int col = DH * h + nt * 16 + (lane & 15);
      const int mb  = (lane >> 4) << 3;
#pragma unroll
      for (int v = 0; v < 8; ++v) {
        C32[(mb + v) * DD + col]      = acc0[v];
        C32[(16 + mb + v) * DD + col] = acc1[v];
      }
    }
    __syncthreads();

    // O f32 -> f16 into A16 cols 0..127 (qkv no longer needed)
    for (int idx = tid; idx < JP * DD; idx += 128) {
      const int m = idx >> 7, n = idx & 127;
      A16[m * QKV + n] = (_Float16)((m < J) ? C32[idx] : 0.0f);
    }
    __syncthreads();

    // attention output projection; then x = LN(x + attn)
    gemm_xwt<false>(A16, P.outW16[l], DD, DD, C32, DD, P.outb[l], wave, lane);
    __syncthreads();
    layernorm<true, false, false>(C32, XRES, nullptr, P.ln1g[l], P.ln1b[l], XRES, A16, wave, lane);
    __syncthreads();

    // FF1 (relu fused) -> C32 (32 x 256)
    gemm_xwt<true>(A16, P.ff1W16[l], DD, FFD, C32, FFD, P.ff1b[l], wave, lane);
    __syncthreads();
    for (int idx = tid; idx < JP * FFD; idx += 128) {
      const int m = idx >> 8, n = idx & 255;
      A16[m * QKV + n] = (_Float16)((m < J) ? C32[idx] : 0.0f);
    }
    __syncthreads();
    // FF2: (28,256)@(256,128); then x = LN(x + ff)
    gemm_xwt<false>(A16, P.ff2W16[l], FFD, DD, C32, DD, P.ff2b[l], wave, lane);
    __syncthreads();
    layernorm<true, false, false>(C32, XRES, nullptr, P.ln2g[l], P.ln2b[l], XRES, A16, wave, lane);
    __syncthreads();
  }

  // ---- final LN ----
  layernorm<false, false, false>(XRES, nullptr, nullptr, P.flng, P.flnb, XRES, A16, wave, lane);
  __syncthreads();

  // ---- decoder: relu(x@W1^T+b1) @ W2^T + b2 ----
  gemm_xwt<true>(A16, P.dW1_16, DD, DD, C32, DD, P.db1, wave, lane);
  __syncthreads();
  if (tid < J * OUTD) {
    const int m = tid / OUTD, o = tid % OUTD;
    float s = P.db2[o];
    for (int k = 0; k < DD; ++k) s += C32[m * DD + k] * P.dW2[o * DD + k];
    P.out[b * (J * OUTD) + tid] = s;
  }
}

// ---- workspace layout (halfs) ----
static const int OFF_G2    = 0;        // 128*128
static const int OFF_INW0  = 16384;    // 384*128
static const int OFF_OUTW0 = 65536;    // 128*128
static const int OFF_FF10  = 81920;    // 256*128
static const int OFF_FF20  = 114688;   // 128*256
static const int OFF_INW1  = 147456;
static const int OFF_OUTW1 = 196608;
static const int OFF_FF11  = 212992;
static const int OFF_FF21  = 245760;
static const int OFF_DW1   = 278528;   // end = 294912 halfs = 576 KiB

extern "C" void kernel_launch(void* const* d_in, const int* in_sizes, int n_in,
                              void* d_out, int out_size, void* d_ws, size_t ws_size,
                              hipStream_t stream) {
  _Float16* w16 = (_Float16*)d_ws;

  // Input index map (setup_inputs flatten order):
  // 0 x, 1 edge_index,
  // 2 g1.W 3 g1.b 4 g1.edge_w 5 g1.lng 6 g1.lnb 7 g1.resW 8 g1.resb,
  // 9 g2.W 10 g2.b 11 g2.edge_w 12 g2.lng 13 g2.lnb,
  // 14 pos, 15 routing,
  // layer l (base 16 + 12*l): inW, inb, outW, outb, ln1g, ln1b, ff1W, ff1b, ff2W, ff2b, ln2g, ln2b
  // 40 flng 41 flnb 42 decW1 43 decb1 44 decW2 45 decb2
  struct Cvt { int idx; int off; int n; };
  const Cvt cv[10] = {
    {9,  OFF_G2,    128 * 128},
    {16, OFF_INW0,  384 * 128}, {18, OFF_OUTW0, 128 * 128},
    {22, OFF_FF10,  256 * 128}, {24, OFF_FF20,  128 * 256},
    {28, OFF_INW1,  384 * 128}, {30, OFF_OUTW1, 128 * 128},
    {32, OFF_FF11,  256 * 128}, {34, OFF_FF21,  128 * 256},
    {42, OFF_DW1,   128 * 128},
  };
  for (int i = 0; i < 10; ++i) {
    f32_to_f16_kernel<<<(cv[i].n + 127) / 128, 128, 0, stream>>>(
        (const float*)d_in[cv[i].idx], w16 + cv[i].off, cv[i].n);
  }

  Params P;
  P.x     = (const float*)d_in[0];
  P.g1W   = (const float*)d_in[2];  P.g1b  = (const float*)d_in[3];
  P.g1lng = (const float*)d_in[5];  P.g1lnb = (const float*)d_in[6];
  P.rW    = (const float*)d_in[7];  P.rb   = (const float*)d_in[8];
  P.g2W16 = w16 + OFF_G2;
  P.g2b   = (const float*)d_in[10];
  P.g2lng = (const float*)d_in[12]; P.g2lnb = (const float*)d_in[13];
  P.pos   = (const float*)d_in[14]; P.rout = (const float*)d_in[15];
  const int iw[2] = {OFF_INW0, OFF_INW1}, ow[2] = {OFF_OUTW0, OFF_OUTW1};
  const int f1[2] = {OFF_FF10, OFF_FF11}, f2[2] = {OFF_FF20, OFF_FF21};
  for (int l = 0; l < 2; ++l) {
    const int base = 16 + 12 * l;
    P.inW16[l]  = w16 + iw[l];  P.inb[l]  = (const float*)d_in[base + 1];
    P.outW16[l] = w16 + ow[l];  P.outb[l] = (const float*)d_in[base + 3];
    P.ln1g[l] = (const float*)d_in[base + 4];  P.ln1b[l] = (const float*)d_in[base + 5];
    P.ff1W16[l] = w16 + f1[l];  P.ff1b[l] = (const float*)d_in[base + 7];
    P.ff2W16[l] = w16 + f2[l];  P.ff2b[l] = (const float*)d_in[base + 9];
    P.ln2g[l] = (const float*)d_in[base + 10]; P.ln2b[l] = (const float*)d_in[base + 11];
  }
  P.flng = (const float*)d_in[40]; P.flnb = (const float*)d_in[41];
  P.dW1_16 = w16 + OFF_DW1;
  P.db1 = (const float*)d_in[43];
  P.dW2 = (const float*)d_in[44]; P.db2 = (const float*)d_in[45];
  P.out = (float*)d_out;

  const int Bn = in_sizes[0] / (J * 2);   // 8192
  sdgrat_fused<<<Bn, 128, 0, stream>>>(P);
}
